// ScoreTransformer_1047972020978
// MI455X (gfx1250) — compile-verified, matrix operand order checked by reading, not run on hardware
//
#include <hip/hip_runtime.h>
#include <hip/hip_bf16.h>
#include <math.h>

// ---------------- problem constants ----------------
constexpr int cB = 32, cN = 128, cD = 512, cFF = 2048, cL = 8, cH = 16, cK = 128;
constexpr int cHD = cD / cH;            // 32
constexpr int c3D = 3 * cD;             // 1536
constexpr int ATOM = 9;
#define SQRT_2PI 2.5066256f             // sqrt(2*3.14159) per reference
#define SCALING  0.17677669529663687f   // 32^-0.5

// ---------------- WMMA types / helpers ----------------
typedef __attribute__((ext_vector_type(16))) __bf16 v16bf;
typedef __attribute__((ext_vector_type(8)))  float  v8f;

__device__ __forceinline__ v8f zero8() {
    v8f z;
    #pragma unroll
    for (int i = 0; i < 8; ++i) z[i] = 0.0f;
    return z;
}

// native f32 -> bf16 convert (hardware v_cvt, RNE)
__device__ __forceinline__ __bf16 f2bf(float f) {
    return (__bf16)f;
}

__device__ __forceinline__ v8f wmma_bf16(v16bf a, v16bf b, v8f c) {
    // D = A(16x32 bf16) * B(32x16 bf16) + C(16x16 f32)
    return __builtin_amdgcn_wmma_f32_16x16x32_bf16(false, a, false, b,
                                                   (short)0, c, false, false);
}

__device__ __forceinline__ float gelu_tanh(float x) {
    float x3 = x * x * x;
    return 0.5f * x * (1.0f + tanhf(0.7978845608028654f * (x + 0.044715f * x3)));
}

// ---------------- geometry ----------------
__global__ void geom_kernel(const float* __restrict__ x,
                            float* __restrict__ dist, float* __restrict__ dpos) {
    int idx = blockIdx.x * blockDim.x + threadIdx.x;       // B*N*N
    if (idx >= cB * cN * cN) return;
    int m = idx % cN, n = (idx / cN) % cN, b = idx / (cN * cN);
    float dx = x[(b * cN + n) * 3 + 0] - x[(b * cN + m) * 3 + 0];
    float dy = x[(b * cN + n) * 3 + 1] - x[(b * cN + m) * 3 + 1];
    float dz = x[(b * cN + n) * 3 + 2] - x[(b * cN + m) * 3 + 2];
    float d = sqrtf(dx * dx + dy * dy + dz * dz);
    dist[idx] = d;
    float inv = 1.0f / (d + 1.0f);
    dpos[idx * 3 + 0] = dx * inv;
    dpos[idx * 3 + 1] = dy * inv;
    dpos[idx * 3 + 2] = dz * inv;
}

// ---------------- Gaussian + bias MLP (K->K silu ->H), WMMA x2 ----------------
__global__ __launch_bounds__(128) void gauss_bias_kernel(
    const float* __restrict__ dist, const int* __restrict__ Z,
    const float* __restrict__ means, const float* __restrict__ stds,
    const float* __restrict__ mul, const float* __restrict__ gbias,
    const float* __restrict__ w1, const float* __restrict__ b1,
    const float* __restrict__ w2, const float* __restrict__ b2,
    float* __restrict__ attn_bias) {
    __shared__ float lds[4][16][cK];
    int lane = threadIdx.x & 31, wave = threadIdx.x >> 5;
    int half = lane >> 4, l16 = lane & 15;
    int tile = blockIdx.x * 4 + wave;                      // 32768 pair-tiles
    int pair = tile * 16 + l16;
    int m = pair % cN, n = (pair / cN) % cN, b = pair / (cN * cN);
    int et = Z[b * cN + n] * ATOM + Z[b * cN + m];
    float xk = mul[et] * dist[pair] + gbias[et];

    // g row (this lane's 64 of 128 K values) directly in WMMA-A layout
    v16bf ga[4];
    #pragma unroll
    for (int t = 0; t < 4; ++t) {
        int ka = t * 32 + half * 8, kb = t * 32 + 16 + half * 8;
        #pragma unroll
        for (int j = 0; j < 8; ++j) {
            float sa = fabsf(stds[ka + j]) + 1e-5f;
            float aa = (xk - means[ka + j]) / sa;
            ga[t][j] = f2bf(__expf(-0.5f * aa * aa) / (SQRT_2PI * sa));
            float sb = fabsf(stds[kb + j]) + 1e-5f;
            float ab = (xk - means[kb + j]) / sb;
            ga[t][8 + j] = f2bf(__expf(-0.5f * ab * ab) / (SQRT_2PI * sb));
        }
    }

    // hidden = silu(g @ w1 + b1)  (16x128), staged to LDS for layout transpose
    float* myl = &lds[wave][0][0];
    for (int jt = 0; jt < 8; ++jt) {
        int colB = jt * 16 + l16;
        v8f acc = zero8();
        #pragma unroll
        for (int t = 0; t < 4; ++t) {
            int ka = t * 32 + half * 8, kb = t * 32 + 16 + half * 8;
            v16bf bv;
            #pragma unroll
            for (int j = 0; j < 8; ++j) {
                bv[j]     = f2bf(w1[(ka + j) * cK + colB]);
                bv[8 + j] = f2bf(w1[(kb + j) * cK + colB]);
            }
            acc = wmma_bf16(ga[t], bv, acc);
        }
        #pragma unroll
        for (int r = 0; r < 8; ++r) {
            float v = acc[r] + b1[colB];
            v = v / (1.0f + __expf(-v));                    // silu
            myl[(r + half * 8) * cK + colB] = v;
        }
    }
    __syncthreads();

    // out = hidden @ w2 + b2  -> (16 pairs x 16 heads)
    v8f acc = zero8();
    #pragma unroll
    for (int t = 0; t < 4; ++t) {
        int ka = t * 32 + half * 8, kb = t * 32 + 16 + half * 8;
        v16bf av, bv;
        #pragma unroll
        for (int j = 0; j < 8; ++j) {
            av[j]     = f2bf(myl[l16 * cK + ka + j]);
            av[8 + j] = f2bf(myl[l16 * cK + kb + j]);
            bv[j]     = f2bf(w2[(ka + j) * cH + l16]);
            bv[8 + j] = f2bf(w2[(kb + j) * cH + l16]);
        }
        acc = wmma_bf16(av, bv, acc);
    }
    #pragma unroll
    for (int r = 0; r < 8; ++r) {
        int p = tile * 16 + r + half * 8;
        int mm = p % cN, nn = (p / cN) % cN, bb = p / (cN * cN);
        attn_bias[(((size_t)bb * cH + l16) * cN + nn) * cN + mm] = acc[r] + b2[l16];
    }
}

// ---------------- gsum[b,n,k] = sum_m g ----------------
__global__ void gsum_kernel(const float* __restrict__ dist, const int* __restrict__ Z,
                            const float* __restrict__ means, const float* __restrict__ stds,
                            const float* __restrict__ mul, const float* __restrict__ gbias,
                            float* __restrict__ gsum) {
    int idx = blockIdx.x * blockDim.x + threadIdx.x;       // B*N*K
    if (idx >= cB * cN * cK) return;
    int k = idx & (cK - 1), n = (idx >> 7) % cN, b = idx / (cN * cK);
    float mean = means[k];
    float sd = fabsf(stds[k]) + 1e-5f;
    float inv = 1.0f / sd, coef = 1.0f / (SQRT_2PI * sd);
    int zn = Z[b * cN + n] * ATOM;
    float s = 0.0f;
    for (int m = 0; m < cN; ++m) {
        int et = zn + Z[b * cN + m];
        float xk = mul[et] * dist[(b * cN + n) * cN + m] + gbias[et];
        float a = (xk - mean) * inv;
        s += __expf(-0.5f * a * a) * coef;
    }
    gsum[idx] = s;
}

// ---------------- feat = atom_emb[Z] + time_emb + edge_feat ----------------
__global__ void feat_init_kernel(const int* __restrict__ Z, const float* __restrict__ t,
                                 const float* __restrict__ atom_emb,
                                 const float* __restrict__ edge_feat,
                                 float* __restrict__ feat) {
    int idx = blockIdx.x * blockDim.x + threadIdx.x;       // B*N*D
    if (idx >= cB * cN * cD) return;
    int d = idx & (cD - 1), n = (idx >> 9) % cN, b = idx / (cN * cD);
    const float cexp = -9.210340371976184f / 255.0f;       // -log(10000)/(half-1)
    float te;
    if (d < 256) te = sinf(t[b] * __expf(d * cexp) * 256.0f);
    else         te = cosf(t[b] * __expf((d - 256) * cexp) * 256.0f);
    feat[idx] = atom_emb[Z[b * cN + n] * cD + d] + te + edge_feat[idx];
}

// ---------------- LayerNorm (one row per block) ----------------
__global__ __launch_bounds__(256) void ln_kernel(const float* __restrict__ X,
                                                 float* __restrict__ Y,
                                                 const float* __restrict__ g,
                                                 const float* __restrict__ b) {
    __shared__ float red[256];
    int row = blockIdx.x, tid = threadIdx.x;
    const float* xr = X + (size_t)row * cD;
    float v0 = xr[tid], v1 = xr[tid + 256];
    red[tid] = v0 + v1; __syncthreads();
    for (int off = 128; off; off >>= 1) { if (tid < off) red[tid] += red[tid + off]; __syncthreads(); }
    float mean = red[0] / (float)cD; __syncthreads();
    float d0 = v0 - mean, d1 = v1 - mean;
    red[tid] = d0 * d0 + d1 * d1; __syncthreads();
    for (int off = 128; off; off >>= 1) { if (tid < off) red[tid] += red[tid + off]; __syncthreads(); }
    float rstd = rsqrtf(red[0] / (float)cD + 1e-5f);
    Y[(size_t)row * cD + tid]       = d0 * rstd * g[tid]       + b[tid];
    Y[(size_t)row * cD + tid + 256] = d1 * rstd * g[tid + 256] + b[tid + 256];
}

// ---------------- generic WMMA GEMM: C = A(MxK) * B(KxN) + bias [+ epi] ----------------
enum { EPI_BIAS = 0, EPI_GELU = 1, EPI_RES = 2 };

template <int EPI>
__global__ __launch_bounds__(128) void gemm_wmma_kernel(
    const float* __restrict__ A, const float* __restrict__ Bm,
    const float* __restrict__ bias, const float* __restrict__ res,
    float* __restrict__ C, int M, int N, int K) {
    int lane = threadIdx.x & 31, wave = threadIdx.x >> 5;
    int half = lane >> 4, l16 = lane & 15;
    int tilesN = N >> 4;
    int tile = blockIdx.x * 4 + wave;
    if (tile >= (M >> 4) * tilesN) return;
    int tM = tile / tilesN, tN = tile % tilesN;

    const float* Arow = A + (size_t)(tM * 16 + l16) * K;
    int colB = tN * 16 + l16;
    v8f acc = zero8();
    for (int k0 = 0; k0 < K; k0 += 32) {
        __builtin_prefetch(Arow + k0 + 32, 0, 1);
        int ka = k0 + half * 8, kb = k0 + 16 + half * 8;
        v16bf av, bv;
        #pragma unroll
        for (int j = 0; j < 8; ++j) {
            av[j]     = f2bf(Arow[ka + j]);
            av[8 + j] = f2bf(Arow[kb + j]);
            bv[j]     = f2bf(Bm[(size_t)(ka + j) * N + colB]);
            bv[8 + j] = f2bf(Bm[(size_t)(kb + j) * N + colB]);
        }
        acc = wmma_bf16(av, bv, acc);
    }
    #pragma unroll
    for (int r = 0; r < 8; ++r) {
        int row = tM * 16 + r + half * 8;
        int col = tN * 16 + l16;
        float v = acc[r] + bias[col];
        if constexpr (EPI == EPI_GELU) v = gelu_tanh(v);
        if constexpr (EPI == EPI_RES)  v += res[(size_t)row * N + col];
        C[(size_t)row * N + col] = v;
    }
}

// ---------------- fused attention (WMMA QK^T + softmax + WMMA PV) ----------------
__global__ __launch_bounds__(128) void attn_kernel(const float* __restrict__ qkv,
                                                   const float* __restrict__ bias,
                                                   float* __restrict__ o) {
    __shared__ float lds[4][16][cN];
    int lane = threadIdx.x & 31, wave = threadIdx.x >> 5;
    int half = lane >> 4, l16 = lane & 15;
    int wid = blockIdx.x * 4 + wave;                       // B*H*(N/16) = 4096
    int nt = wid & 7, h = (wid >> 3) & 15, b = wid >> 7;

    int nrow = nt * 16 + l16;
    const float* qrow = qkv + (size_t)(b * cN + nrow) * c3D + h * cHD;
    v16bf aq;
    {
        int ka = half * 8, kb = 16 + half * 8;
        #pragma unroll
        for (int j = 0; j < 8; ++j) {
            aq[j]     = f2bf(qrow[ka + j] * SCALING);
            aq[8 + j] = f2bf(qrow[kb + j] * SCALING);
        }
    }
    v8f sc[8];
    for (int jt = 0; jt < 8; ++jt) {
        int mcol = jt * 16 + l16;
        const float* krow = qkv + (size_t)(b * cN + mcol) * c3D + cD + h * cHD;
        v16bf bk;
        int ka = half * 8, kb = 16 + half * 8;
        #pragma unroll
        for (int j = 0; j < 8; ++j) {
            bk[j]     = f2bf(krow[ka + j]);
            bk[8 + j] = f2bf(krow[kb + j]);
        }
        sc[jt] = wmma_bf16(aq, bk, zero8());
    }
    // additive bias + row softmax (rows live in (r, half); cols across 16-lane halves)
    const float* bptr = bias + ((size_t)b * cH + h) * cN * cN;
    #pragma unroll
    for (int r = 0; r < 8; ++r) {
        int row = nt * 16 + r + half * 8;
        float mx = -1e30f;
        #pragma unroll
        for (int jt = 0; jt < 8; ++jt) {
            sc[jt][r] += bptr[(size_t)row * cN + jt * 16 + l16];
            mx = fmaxf(mx, sc[jt][r]);
        }
        for (int off = 1; off < 16; off <<= 1) mx = fmaxf(mx, __shfl_xor(mx, off, 32));
        float s = 0.0f;
        #pragma unroll
        for (int jt = 0; jt < 8; ++jt) { sc[jt][r] = __expf(sc[jt][r] - mx); s += sc[jt][r]; }
        for (int off = 1; off < 16; off <<= 1) s += __shfl_xor(s, off, 32);
        float inv = 1.0f / s;
        #pragma unroll
        for (int jt = 0; jt < 8; ++jt) sc[jt][r] *= inv;
    }
    // transpose P through LDS into WMMA-A layout
    float* myl = &lds[wave][0][0];
    #pragma unroll
    for (int r = 0; r < 8; ++r) {
        int rl = r + half * 8;
        #pragma unroll
        for (int jt = 0; jt < 8; ++jt) myl[rl * cN + jt * 16 + l16] = sc[jt][r];
    }
    __syncthreads();
    // O = P(16x128) @ V(128x32)
    for (int jo = 0; jo < 2; ++jo) {
        int kd = jo * 16 + l16;
        v8f acc = zero8();
        #pragma unroll
        for (int t = 0; t < 4; ++t) {
            int ka = t * 32 + half * 8, kb = t * 32 + 16 + half * 8;
            v16bf av, bv;
            #pragma unroll
            for (int j = 0; j < 8; ++j) {
                av[j]     = f2bf(myl[l16 * cN + ka + j]);
                av[8 + j] = f2bf(myl[l16 * cN + kb + j]);
                bv[j]     = f2bf(qkv[(size_t)(b * cN + ka + j) * c3D + 2 * cD + h * cHD + kd]);
                bv[8 + j] = f2bf(qkv[(size_t)(b * cN + kb + j) * c3D + 2 * cD + h * cHD + kd]);
            }
            acc = wmma_bf16(av, bv, acc);
        }
        #pragma unroll
        for (int r = 0; r < 8; ++r) {
            int row = nt * 16 + r + half * 8;
            o[(size_t)(b * cN + row) * cD + h * cHD + jo * 16 + l16] = acc[r];
        }
    }
}

// ---------------- vw[b,n,h] = sum_d v[b,n,h,d] * force_w[h*HD+d] ----------------
__global__ void vw_kernel(const float* __restrict__ vh, const float* __restrict__ fw,
                          float* __restrict__ vw) {
    int idx = blockIdx.x * blockDim.x + threadIdx.x;       // B*N*H
    if (idx >= cB * cN * cH) return;
    int h = idx & (cH - 1), bn = idx >> 4;
    float s = 0.0f;
    #pragma unroll
    for (int d = 0; d < cHD; ++d) s += vh[(size_t)bn * cD + h * cHD + d] * fw[h * cHD + d];
    vw[idx] = s;
}

// ---------------- NodeTaskHead: WMMA scores, softmax, contract with vw & dpos ----------------
__global__ __launch_bounds__(128) void head_kernel(const float* __restrict__ qh,
                                                   const float* __restrict__ kh,
                                                   const float* __restrict__ bias,
                                                   const float* __restrict__ vw,
                                                   const float* __restrict__ dpos,
                                                   float* __restrict__ forces) {
    int lane = threadIdx.x & 31, wave = threadIdx.x >> 5;
    int half = lane >> 4, l16 = lane & 15;
    int wid = blockIdx.x * 4 + wave;                       // B*(N/16) = 256
    int nt = wid & 7, b = wid >> 3;

    float sacc[8][8];
    #pragma unroll
    for (int jt = 0; jt < 8; ++jt) {
        #pragma unroll
        for (int r = 0; r < 8; ++r) sacc[jt][r] = 0.0f;
    }

    for (int h = 0; h < cH; ++h) {
        int nrow = nt * 16 + l16;
        const float* qrow = qh + (size_t)(b * cN + nrow) * cD + h * cHD;
        v16bf aq;
        {
            int ka = half * 8, kb = 16 + half * 8;
            #pragma unroll
            for (int j = 0; j < 8; ++j) {
                aq[j]     = f2bf(qrow[ka + j] * SCALING);
                aq[8 + j] = f2bf(qrow[kb + j] * SCALING);
            }
        }
        v8f sc[8];
        for (int jt = 0; jt < 8; ++jt) {
            int mcol = jt * 16 + l16;
            const float* krow = kh + (size_t)(b * cN + mcol) * cD + h * cHD;
            v16bf bk;
            int ka = half * 8, kb = 16 + half * 8;
            #pragma unroll
            for (int j = 0; j < 8; ++j) {
                bk[j]     = f2bf(krow[ka + j]);
                bk[8 + j] = f2bf(krow[kb + j]);
            }
            sc[jt] = wmma_bf16(aq, bk, zero8());
        }
        const float* bptr = bias + ((size_t)b * cH + h) * cN * cN;
        #pragma unroll
        for (int r = 0; r < 8; ++r) {
            int row = nt * 16 + r + half * 8;
            float mx = -1e30f;
            #pragma unroll
            for (int jt = 0; jt < 8; ++jt) {
                sc[jt][r] += bptr[(size_t)row * cN + jt * 16 + l16];
                mx = fmaxf(mx, sc[jt][r]);
            }
            for (int off = 1; off < 16; off <<= 1) mx = fmaxf(mx, __shfl_xor(mx, off, 32));
            float s = 0.0f;
            #pragma unroll
            for (int jt = 0; jt < 8; ++jt) { sc[jt][r] = __expf(sc[jt][r] - mx); s += sc[jt][r]; }
            for (int off = 1; off < 16; off <<= 1) s += __shfl_xor(s, off, 32);
            float inv = 1.0f / s;
            #pragma unroll
            for (int jt = 0; jt < 8; ++jt)
                sacc[jt][r] += sc[jt][r] * inv * vw[(size_t)(b * cN + jt * 16 + l16) * cH + h];
        }
    }
    // forces[b,n,c] = sum_m sacc[n,m] * dpos[b,n,m,c]
    #pragma unroll
    for (int r = 0; r < 8; ++r) {
        int row = nt * 16 + r + half * 8;
        float f0 = 0.0f, f1 = 0.0f, f2 = 0.0f;
        #pragma unroll
        for (int jt = 0; jt < 8; ++jt) {
            int m = jt * 16 + l16;
            const float* dp = dpos + ((size_t)(b * cN + row) * cN + m) * 3;
            f0 += sacc[jt][r] * dp[0];
            f1 += sacc[jt][r] * dp[1];
            f2 += sacc[jt][r] * dp[2];
        }
        for (int off = 1; off < 16; off <<= 1) {
            f0 += __shfl_xor(f0, off, 32);
            f1 += __shfl_xor(f1, off, 32);
            f2 += __shfl_xor(f2, off, 32);
        }
        if (l16 == 0) {
            forces[(size_t)(b * cN + row) * 3 + 0] = f0;
            forces[(size_t)(b * cN + row) * 3 + 1] = f1;
            forces[(size_t)(b * cN + row) * 3 + 2] = f2;
        }
    }
}

// ---------------- host launch ----------------
extern "C" void kernel_launch(void* const* d_in, const int* in_sizes, int n_in,
                              void* d_out, int out_size, void* d_ws, size_t ws_size,
                              hipStream_t stream) {
    (void)in_sizes; (void)n_in; (void)out_size; (void)ws_size;
    const float* x        = (const float*)d_in[0];
    const int*   Z        = (const int*)  d_in[1];
    const float* t        = (const float*)d_in[2];
    const float* atom_emb = (const float*)d_in[3];
    const float* ln1_g    = (const float*)d_in[4];
    const float* ln1_b    = (const float*)d_in[5];
    const float* in_w     = (const float*)d_in[6];
    const float* in_b     = (const float*)d_in[7];
    const float* out_w    = (const float*)d_in[8];
    const float* out_b    = (const float*)d_in[9];
    const float* ffn_ln_g = (const float*)d_in[10];
    const float* ffn_ln_b = (const float*)d_in[11];
    const float* ffn_w1   = (const float*)d_in[12];
    const float* ffn_b1   = (const float*)d_in[13];
    const float* ffn_w2   = (const float*)d_in[14];
    const float* ffn_b2   = (const float*)d_in[15];
    const float* fln_g    = (const float*)d_in[16];
    const float* fln_b    = (const float*)d_in[17];
    const float* gbf_means= (const float*)d_in[18];
    const float* gbf_stds = (const float*)d_in[19];
    const float* gbf_mul  = (const float*)d_in[20];
    const float* gbf_bias = (const float*)d_in[21];
    const float* bp_w1    = (const float*)d_in[22];
    const float* bp_b1    = (const float*)d_in[23];
    const float* bp_w2    = (const float*)d_in[24];
    const float* bp_b2    = (const float*)d_in[25];
    const float* edge_w   = (const float*)d_in[26];
    const float* edge_b   = (const float*)d_in[27];
    const float* q_w      = (const float*)d_in[28];
    const float* q_b      = (const float*)d_in[29];
    const float* k_w      = (const float*)d_in[30];
    const float* k_b      = (const float*)d_in[31];
    const float* v_w      = (const float*)d_in[32];
    const float* v_b      = (const float*)d_in[33];
    const float* force_w  = (const float*)d_in[34];
    float* forces = (float*)d_out;

    float* ws = (float*)d_ws;
    const size_t PAIRS = (size_t)cB * cN * cN;             // 524288
    const size_t ROWS  = (size_t)cB * cN;                  // 4096
    float* dist   = ws;                 size_t off = PAIRS;
    float* dpos   = ws + off;           off += PAIRS * 3;
    float* abias  = ws + off;           off += (size_t)cB * cH * cN * cN;
    float* gsum   = ws + off;           off += ROWS * cK;
    float* feat   = ws + off;           off += ROWS * cD;
    float* hbuf   = ws + off;           off += ROWS * cD;
    float* qkv    = ws + off;           off += ROWS * c3D;
    float* obuf   = ws + off;           off += ROWS * cD;   // also edge_feat
    float* hid    = ws + off;           off += ROWS * cFF;
    float* featn  = ws + off;           off += ROWS * cD;
    float* vwbuf  = ws + off;           off += ROWS * cH;
    float* qh = qkv;                                        // reuse qkv space
    float* kh = qkv + ROWS * cD;
    float* vh = qkv + 2 * ROWS * cD;

    auto gemm_blocks = [](int M, int N) { return ((M >> 4) * (N >> 4) + 3) / 4; };

    // geometry + gaussian bias + edge features
    geom_kernel<<<(int)((PAIRS + 255) / 256), 256, 0, stream>>>(x, dist, dpos);
    gauss_bias_kernel<<<(int)(PAIRS / 16 / 4), 128, 0, stream>>>(
        dist, Z, gbf_means, gbf_stds, gbf_mul, gbf_bias,
        bp_w1, bp_b1, bp_w2, bp_b2, abias);
    gsum_kernel<<<(int)((ROWS * cK + 255) / 256), 256, 0, stream>>>(
        dist, Z, gbf_means, gbf_stds, gbf_mul, gbf_bias, gsum);
    gemm_wmma_kernel<EPI_BIAS><<<gemm_blocks(ROWS, cD), 128, 0, stream>>>(
        gsum, edge_w, edge_b, nullptr, obuf, (int)ROWS, cD, cK);
    feat_init_kernel<<<(int)((ROWS * cD + 255) / 256), 256, 0, stream>>>(
        Z, t, atom_emb, obuf, feat);

    // transformer layers
    for (int i = 0; i < cL; ++i) {
        ln_kernel<<<(int)ROWS, 256, 0, stream>>>(feat, hbuf, ln1_g + i * cD, ln1_b + i * cD);
        gemm_wmma_kernel<EPI_BIAS><<<gemm_blocks(ROWS, c3D), 128, 0, stream>>>(
            hbuf, in_w + (size_t)i * cD * c3D, in_b + i * c3D, nullptr,
            qkv, (int)ROWS, c3D, cD);
        attn_kernel<<<(int)(cB * cH * (cN / 16) / 4), 128, 0, stream>>>(qkv, abias, obuf);
        gemm_wmma_kernel<EPI_RES><<<gemm_blocks(ROWS, cD), 128, 0, stream>>>(
            obuf, out_w + (size_t)i * cD * cD, out_b + i * cD, feat,
            feat, (int)ROWS, cD, cD);
        ln_kernel<<<(int)ROWS, 256, 0, stream>>>(feat, hbuf, ffn_ln_g + i * cD, ffn_ln_b + i * cD);
        gemm_wmma_kernel<EPI_GELU><<<gemm_blocks(ROWS, cFF), 128, 0, stream>>>(
            hbuf, ffn_w1 + (size_t)i * cD * cFF, ffn_b1 + i * cFF, nullptr,
            hid, (int)ROWS, cFF, cD);
        gemm_wmma_kernel<EPI_RES><<<gemm_blocks(ROWS, cD), 128, 0, stream>>>(
            hid, ffn_w2 + (size_t)i * cFF * cD, ffn_b2 + i * cD, feat,
            feat, (int)ROWS, cD, cFF);
    }

    // final LN + head
    ln_kernel<<<(int)ROWS, 256, 0, stream>>>(feat, featn, fln_g, fln_b);
    gemm_wmma_kernel<EPI_BIAS><<<gemm_blocks(ROWS, cD), 128, 0, stream>>>(
        featn, q_w, q_b, nullptr, qh, (int)ROWS, cD, cD);
    gemm_wmma_kernel<EPI_BIAS><<<gemm_blocks(ROWS, cD), 128, 0, stream>>>(
        featn, k_w, k_b, nullptr, kh, (int)ROWS, cD, cD);
    gemm_wmma_kernel<EPI_BIAS><<<gemm_blocks(ROWS, cD), 128, 0, stream>>>(
        featn, v_w, v_b, nullptr, vh, (int)ROWS, cD, cD);
    vw_kernel<<<(int)((ROWS * cH + 255) / 256), 256, 0, stream>>>(vh, force_w, vwbuf);
    head_kernel<<<(int)(cB * (cN / 16) / 4), 128, 0, stream>>>(
        qh, kh, abias, vwbuf, dpos, forces);
}
